// MiniLCX_7980049236541
// MI455X (gfx1250) — matrix operand order repaired and checked by reading, not compiled, not run on hardware
//
#include <hip/hip_runtime.h>
#include <hip/hip_bf16.h>
#include <math.h>

// ---------------------------------------------------------------------------
// Problem sizes (fixed by the reference)
// ---------------------------------------------------------------------------
#define NROW 8192     // hidden rows
#define DH   1024     // hidden dim
#define KD   256      // key dim
#define SK   32768    // number of keys/values
#define TOPK 8

typedef __bf16 bf16_t;
typedef __attribute__((ext_vector_type(16))) __bf16 v16bf;
typedef __attribute__((ext_vector_type(8)))  __bf16 v8bf;
typedef __attribute__((ext_vector_type(8)))  float  v8f;

static __device__ __forceinline__ v16bf join16(v8bf a, v8bf b) {
  return __builtin_shufflevector(a, b, 0,1,2,3,4,5,6,7,8,9,10,11,12,13,14,15);
}

// ---------------------------------------------------------------------------
// P_valid: pack valid[] bools into 1024 bitmask words + compute any(valid)
// ---------------------------------------------------------------------------
__global__ __launch_bounds__(1024) void k_validbits(const unsigned char* __restrict__ valid,
                                                    unsigned* __restrict__ bits,
                                                    int* __restrict__ anyvalid) {
  __shared__ unsigned red[1024];
  int t = threadIdx.x;
  unsigned w = 0u;
  #pragma unroll 4
  for (int j = 0; j < 32; ++j) w |= (valid[t * 32 + j] ? 1u : 0u) << j;
  bits[t] = w;
  red[t] = w;
  __syncthreads();
  for (int s = 512; s > 0; s >>= 1) {
    if (t < s) red[t] |= red[t + s];
    __syncthreads();
  }
  if (t == 0) *anyvalid = (red[0] != 0u) ? 1 : 0;
}

// ---------------------------------------------------------------------------
// P_keys: k = normalize(keys) rows, stored as bf16 hi + bf16 residual lo
// ---------------------------------------------------------------------------
__global__ __launch_bounds__(256) void k_keynorm(const float* __restrict__ keys,
                                                 bf16_t* __restrict__ kn_hi,
                                                 bf16_t* __restrict__ kn_lo) {
  __shared__ float red[256];
  int r = blockIdx.x, t = threadIdx.x;
  float x = keys[(size_t)r * KD + t];
  red[t] = x * x;
  __syncthreads();
  for (int s = 128; s > 0; s >>= 1) {
    if (t < s) red[t] += red[t + s];
    __syncthreads();
  }
  float inv = 1.0f / fmaxf(sqrtf(red[0]), 1e-12f);
  float y = x * inv;
  __bf16 h = (__bf16)y;
  kn_hi[(size_t)r * KD + t] = h;
  kn_lo[(size_t)r * KD + t] = (__bf16)(y - (float)h);
}

// ---------------------------------------------------------------------------
// P_gate: per-row sigmoid(hidden . w_g + b_g)
// ---------------------------------------------------------------------------
__global__ __launch_bounds__(256) void k_gate(const float* __restrict__ hidden,
                                              const float* __restrict__ w_g,
                                              const float* __restrict__ b_g,
                                              float* __restrict__ g_row) {
  __shared__ float red[256];
  int r = blockIdx.x, t = threadIdx.x;
  float s = 0.f;
  #pragma unroll
  for (int u = 0; u < 4; ++u) {
    int c = t + u * 256;
    s += hidden[(size_t)r * DH + c] * w_g[c];
  }
  red[t] = s;
  __syncthreads();
  for (int k = 128; k > 0; k >>= 1) {
    if (t < k) red[t] += red[t + k];
    __syncthreads();
  }
  if (t == 0) g_row[r] = 1.0f / (1.0f + __expf(-(red[0] + b_g[0])));
}

// ---------------------------------------------------------------------------
// P_colsum: partial column sums of hidden (for write_val), 64 blocks x 128 rows
// ---------------------------------------------------------------------------
__global__ __launch_bounds__(256) void k_colsum(const float* __restrict__ hidden,
                                                float* __restrict__ part) {
  int b = blockIdx.x, t = threadIdx.x;
  float acc[4] = {0.f, 0.f, 0.f, 0.f};
  for (int r = 0; r < 128; ++r) {
    size_t base = (size_t)(b * 128 + r) * DH;
    #pragma unroll
    for (int u = 0; u < 4; ++u) acc[u] += hidden[base + t + u * 256];
  }
  #pragma unroll
  for (int u = 0; u < 4; ++u) part[(size_t)b * DH + t + u * 256] = acc[u];
}

// ---------------------------------------------------------------------------
// Q projection: q_pre = hidden @ w_q^T + b_q  (bf16 hi/lo split WMMA, f32 acc)
// then row-normalize -> q_hi/q_lo bf16, plus per-block column sums of q_pre.
// Block = 128 threads (4 waves); block owns 16 rows; wave owns 4 of 16 n-tiles.
// ---------------------------------------------------------------------------
__global__ __launch_bounds__(128) void k_proj(const float* __restrict__ hidden,
                                              const float* __restrict__ w_q,
                                              const float* __restrict__ b_q,
                                              bf16_t* __restrict__ q_hi,
                                              bf16_t* __restrict__ q_lo,
                                              float* __restrict__ qcol_part) {
  __shared__ float qpre[16 * KD];     // 16 KB
  __shared__ float rpart[16][8];
  __shared__ float invn[16];
  int tid = threadIdx.x;
  int wave = tid >> 5, lane = tid & 31;
  int m0 = blockIdx.x * 16;
  int l15 = lane & 15;
  int hi = lane >> 4;

  v8f acc[4];
  #pragma unroll
  for (int nt = 0; nt < 4; ++nt) acc[nt] = (v8f){};

  for (int kc = 0; kc < 32; ++kc) {
    int kbase = kc * 32;
    // A fragment from hidden row m0+l15: K = kbase + hi*8 + {0..7, 16..23}
    const float* ha = hidden + (size_t)(m0 + l15) * DH + kbase + hi * 8;
    v16bf Ah, Al;
    #pragma unroll
    for (int j = 0; j < 8; ++j) {
      float x = ha[j];
      __bf16 h = (__bf16)x;
      Ah[j] = h; Al[j] = (__bf16)(x - (float)h);
      float y = ha[16 + j];
      __bf16 h2 = (__bf16)y;
      Ah[8 + j] = h2; Al[8 + j] = (__bf16)(y - (float)h2);
    }
    #pragma unroll
    for (int nt = 0; nt < 4; ++nt) {
      int ncol = (wave * 4 + nt) * 16 + l15;  // kd index (B column)
      const float* wb = w_q + (size_t)ncol * DH + kbase + hi * 8;
      v16bf Bh, Bl;
      #pragma unroll
      for (int j = 0; j < 16; ++j) {
        float x = wb[j < 8 ? j : j + 8];
        __bf16 h = (__bf16)x;
        Bh[j] = h; Bl[j] = (__bf16)(x - (float)h);
      }
      acc[nt] = __builtin_amdgcn_wmma_f32_16x16x32_bf16(false, Ah, false, Bh, (short)0, acc[nt], false, false);
      acc[nt] = __builtin_amdgcn_wmma_f32_16x16x32_bf16(false, Ah, false, Bl, (short)0, acc[nt], false, false);
      acc[nt] = __builtin_amdgcn_wmma_f32_16x16x32_bf16(false, Al, false, Bh, (short)0, acc[nt], false, false);
    }
  }
  // Stage q_pre (+bias) into LDS.  D element (i,lane): row=i+8*hi, col=l15 of tile.
  #pragma unroll
  for (int nt = 0; nt < 4; ++nt) {
    int c = (wave * 4 + nt) * 16 + l15;
    float bq = b_q[c];
    #pragma unroll
    for (int i = 0; i < 8; ++i) qpre[(i + 8 * hi) * KD + c] = acc[nt][i] + bq;
  }
  __syncthreads();
  // Row norms (16 rows x 8 segments of 32)
  {
    int r = tid >> 3, seg = tid & 7;
    float ss = 0.f;
    for (int j = 0; j < 32; ++j) {
      float x = qpre[r * KD + seg * 32 + j];
      ss += x * x;
    }
    rpart[r][seg] = ss;
  }
  __syncthreads();
  if (tid < 16) {
    float ss = 0.f;
    #pragma unroll
    for (int j = 0; j < 8; ++j) ss += rpart[tid][j];
    invn[tid] = 1.0f / fmaxf(sqrtf(ss), 1e-12f);
  }
  __syncthreads();
  // Per-block column sums of q_pre (pre-normalization) for write_key
  {
    int c0 = tid * 2;
    float s0 = 0.f, s1 = 0.f;
    for (int r = 0; r < 16; ++r) {
      s0 += qpre[r * KD + c0];
      s1 += qpre[r * KD + c0 + 1];
    }
    qcol_part[(size_t)blockIdx.x * KD + c0]     = s0;
    qcol_part[(size_t)blockIdx.x * KD + c0 + 1] = s1;
  }
  // Normalized q in bf16 hi/lo
  for (int r = 0; r < 16; ++r) {
    int c0 = tid * 2;
    #pragma unroll
    for (int u = 0; u < 2; ++u) {
      float y = qpre[r * KD + c0 + u] * invn[r];
      __bf16 h = (__bf16)y;
      q_hi[(size_t)(m0 + r) * KD + c0 + u] = h;
      q_lo[(size_t)(m0 + r) * KD + c0 + u] = (__bf16)(y - (float)h);
    }
  }
}

// ---------------------------------------------------------------------------
// S: fused scores GEMM (bf16 hi/lo split, f32 acc) + streaming top-8.
// Block = 128 threads (4 waves) owns 16 rows; wave w scans S range [w*8192, ..).
// Per 16-col tile: 24 WMMAs on 3 independent accumulator chains -> LDS stage ->
// per-lane sorted top-8 update. Next tile's B lines are prefetched.
// ---------------------------------------------------------------------------
__global__ __launch_bounds__(128) void k_scores_topk(
    const bf16_t* __restrict__ q_hi, const bf16_t* __restrict__ q_lo,
    const bf16_t* __restrict__ kn_hi, const bf16_t* __restrict__ kn_lo,
    const unsigned* __restrict__ validbits, const int* __restrict__ anyvalid,
    float* __restrict__ top_vals, int* __restrict__ top_idx,
    float* __restrict__ out_slot) {
  __shared__ float stage[4 * 256];    // per-wave 16x16 f32 tile
  __shared__ float mv[4 * 32 * TOPK];
  __shared__ int   mi[4 * 32 * TOPK];
  int tid = threadIdx.x, wave = tid >> 5, lane = tid & 31;
  int m0 = blockIdx.x * 16;
  int l15 = lane & 15;
  int hi = lane >> 4;

  // A fragments for all 8 K-chunks (q rows m0..m0+15), kept in registers.
  v16bf Ah[8], Al[8];
  #pragma unroll
  for (int kc = 0; kc < 8; ++kc) {
    const bf16_t* ph = q_hi + (size_t)(m0 + l15) * KD + kc * 32 + hi * 16;
    const bf16_t* pl = q_lo + (size_t)(m0 + l15) * KD + kc * 32 + hi * 16;
    v8bf a0 = *(const v8bf*)(ph);
    v8bf a1 = *(const v8bf*)(ph + 8);
    v8bf b0 = *(const v8bf*)(pl);
    v8bf b1 = *(const v8bf*)(pl + 8);
    Ah[kc] = join16(a0, a1);
    Al[kc] = join16(b0, b1);
  }

  bool allv = (*anyvalid == 0);
  float tv[TOPK];
  int   ti[TOPK];
  #pragma unroll
  for (int j = 0; j < TOPK; ++j) { tv[j] = -INFINITY; ti[j] = 0; }

  int sBegin = wave * (SK / 4);
  float* st = &stage[wave * 256];
  int crow = lane & 15;            // consumed row
  int cbase = (lane >> 4) * 8;     // consumed column half

  for (int t = 0; t < (SK / 4) / 16; ++t) {
    int n0 = sBegin + t * 16;
    // Prefetch next tile's B fragment lines into cache (hides L2 latency
    // behind the WMMA bundle; emits global_prefetch on gfx1250).
    if (t + 1 < (SK / 4) / 16) {
      const bf16_t* nph = kn_hi + (size_t)(n0 + 16 + l15) * KD;
      const bf16_t* npl = kn_lo + (size_t)(n0 + 16 + l15) * KD;
      __builtin_prefetch(nph, 0, 1);
      __builtin_prefetch(npl, 0, 1);
    }
    // Three independent accumulator chains: hi*hi, hi*lo, lo*hi
    v8f c0 = (v8f){}, c1 = (v8f){}, c2 = (v8f){};
    #pragma unroll
    for (int kc = 0; kc < 8; ++kc) {
      const bf16_t* bh = kn_hi + (size_t)(n0 + l15) * KD + kc * 32 + hi * 16;
      const bf16_t* bl = kn_lo + (size_t)(n0 + l15) * KD + kc * 32 + hi * 16;
      v16bf Bh = *(const v16bf*)bh;   // 32B aligned contiguous fragment
      v16bf Bl = *(const v16bf*)bl;
      c0 = __builtin_amdgcn_wmma_f32_16x16x32_bf16(false, Ah[kc], false, Bh, (short)0, c0, false, false);
      c1 = __builtin_amdgcn_wmma_f32_16x16x32_bf16(false, Ah[kc], false, Bl, (short)0, c1, false, false);
      c2 = __builtin_amdgcn_wmma_f32_16x16x32_bf16(false, Al[kc], false, Bh, (short)0, c2, false, false);
    }
    v8f c = (c0 + c1) + c2;
    // Stage the 16x16 tile: element (i,lane) -> row i+8*hi, col l15
    #pragma unroll
    for (int i = 0; i < 8; ++i) st[(i + 8 * hi) * 16 + l15] = c[i];
    __builtin_amdgcn_wave_barrier();   // keep DS order; same-wave LDS is in-order
    unsigned w = validbits[n0 >> 5];
    unsigned mbits = allv ? 0xffffu : ((w >> (n0 & 31)) & 0xffffu);
    const float* rowp = &st[crow * 16];
    #pragma unroll
    for (int j = 0; j < 8; ++j) {
      int cc = cbase + j;
      if ((mbits >> cc) & 1u) {
        float s = rowp[cc];
        if (s > tv[TOPK - 1]) {          // rare path: sorted insert
          int id = n0 + cc;
          #pragma unroll
          for (int p = 0; p < TOPK; ++p) {
            bool g = s > tv[p];
            float ns = g ? s : tv[p]; float os = g ? tv[p] : s;
            int nid = g ? id : ti[p];  int oid = g ? ti[p] : id;
            tv[p] = ns; ti[p] = nid; s = os; id = oid;
          }
        }
      }
    }
    __builtin_amdgcn_wave_barrier();
  }

  // Merge 4 waves x 2 lane-halves per row -> final sorted top-8 per row.
  #pragma unroll
  for (int j = 0; j < TOPK; ++j) {
    mv[(wave * 32 + lane) * TOPK + j] = tv[j];
    mi[(wave * 32 + lane) * TOPK + j] = ti[j];
  }
  __syncthreads();
  if (tid < 16) {
    float fv[TOPK]; int fi[TOPK];
    #pragma unroll
    for (int j = 0; j < TOPK; ++j) { fv[j] = -INFINITY; fi[j] = 0; }
    for (int w = 0; w < 4; ++w) {
      for (int h = 0; h < 2; ++h) {
        int l = w * 32 + h * 16 + tid;
        #pragma unroll
        for (int j = 0; j < TOPK; ++j) {
          float s = mv[l * TOPK + j];
          int id  = mi[l * TOPK + j];
          if (s > fv[TOPK - 1]) {
            #pragma unroll
            for (int p = 0; p < TOPK; ++p) {
              bool g = s > fv[p];
              float ns = g ? s : fv[p]; float os = g ? fv[p] : s;
              int nid = g ? id : fi[p]; int oid = g ? fi[p] : id;
              fv[p] = ns; fi[p] = nid; s = os; id = oid;
            }
          }
        }
      }
    }
    int r = m0 + tid;
    #pragma unroll
    for (int j = 0; j < TOPK; ++j) {
      top_vals[(size_t)r * TOPK + j] = fv[j];
      top_idx[(size_t)r * TOPK + j]  = fi[j];
    }
    out_slot[r] = (float)fi[0];
  }
}

// ---------------------------------------------------------------------------
// C: softmax over top-8 + weighted gather of values rows -> context
// ---------------------------------------------------------------------------
__global__ __launch_bounds__(256) void k_context(const float* __restrict__ top_vals,
                                                 const int* __restrict__ top_idx,
                                                 const float* __restrict__ values,
                                                 float* __restrict__ context) {
  __shared__ float sv[TOPK];
  __shared__ int   si[TOPK];
  int n = blockIdx.x, t = threadIdx.x;
  if (t < TOPK) {
    sv[t] = top_vals[(size_t)n * TOPK + t];
    si[t] = top_idx[(size_t)n * TOPK + t];
  }
  __syncthreads();
  float m = sv[0];
  float e[TOPK], den = 0.f;
  #pragma unroll
  for (int j = 0; j < TOPK; ++j) { e[j] = __expf(sv[j] - m); den += e[j]; }
  float inv = 1.0f / den;
  #pragma unroll
  for (int u = 0; u < 4; ++u) {
    int c = t + u * 256;
    float acc = 0.f;
    #pragma unroll
    for (int j = 0; j < TOPK; ++j)
      acc += (e[j] * inv) * values[(size_t)si[j] * DH + c];
    context[(size_t)n * DH + c] = acc;
  }
}

// ---------------------------------------------------------------------------
// E: deterministic single-block epilogue: margin, gate, slot, write_key/val
// ---------------------------------------------------------------------------
__global__ __launch_bounds__(256) void k_epilogue(
    const float* __restrict__ top_vals, const int* __restrict__ top_idx,
    const float* __restrict__ g_row, const float* __restrict__ qcol_part,
    const float* __restrict__ hcol_part,
    float* __restrict__ out_margin,
    float* __restrict__ ws_gate, int* __restrict__ ws_slot,
    float* __restrict__ ws_wk, float* __restrict__ ws_wv) {
  __shared__ float red[256];
  __shared__ int   ired[256];
  int t = threadIdx.x;

  // margin + inf flag
  float msum = 0.f; int infl = 0;
  for (int r = t; r < NROW; r += 256) {
    float v0 = top_vals[(size_t)r * TOPK + 0];
    float v1 = top_vals[(size_t)r * TOPK + 1];
    msum += v0 - v1;
    #pragma unroll
    for (int j = 0; j < TOPK; ++j)
      infl |= (fabsf(top_vals[(size_t)r * TOPK + j]) == INFINITY) ? 1 : 0;
  }
  red[t] = msum; ired[t] = infl;
  __syncthreads();
  for (int s = 128; s > 0; s >>= 1) {
    if (t < s) { red[t] += red[t + s]; ired[t] |= ired[t + s]; }
    __syncthreads();
  }
  if (t == 0) out_margin[0] = ired[0] ? 0.0f : red[0] / (float)NROW;
  __syncthreads();

  // gate
  float gs = 0.f;
  for (int r = t; r < NROW; r += 256) gs += g_row[r];
  red[t] = gs;
  __syncthreads();
  for (int s = 128; s > 0; s >>= 1) {
    if (t < s) red[t] += red[t + s];
    __syncthreads();
  }
  if (t == 0) { *ws_gate = red[0] / (float)NROW; *ws_slot = top_idx[0]; }
  __syncthreads();

  // write_key = normalize(mean of q_pre columns)
  float qs = 0.f;
  for (int b = 0; b < 512; ++b) qs += qcol_part[(size_t)b * KD + t];
  float wkm = qs / (float)NROW;
  red[t] = wkm * wkm;
  __syncthreads();
  for (int s = 128; s > 0; s >>= 1) {
    if (t < s) red[t] += red[t + s];
    __syncthreads();
  }
  float inv = 1.0f / fmaxf(sqrtf(red[0]), 1e-12f);
  ws_wk[t] = wkm * inv;

  // write_val = column mean of hidden
  #pragma unroll
  for (int u = 0; u < 4; ++u) {
    int c = t + u * 256;
    float hs = 0.f;
    for (int b = 0; b < 64; ++b) hs += hcol_part[(size_t)b * DH + c];
    ws_wv[c] = hs / (float)NROW;
  }
}

// ---------------------------------------------------------------------------
// K/V: full copies with one blended row (bandwidth bound, ~320 MB total)
// ---------------------------------------------------------------------------
__global__ __launch_bounds__(256) void k_newkeys(const float* __restrict__ keys,
                                                 const float* __restrict__ ws_wk,
                                                 const float* __restrict__ ws_gate,
                                                 const int* __restrict__ ws_slot,
                                                 float* __restrict__ out) {
  size_t base = (size_t)blockIdx.x * 1024;
  float g = *ws_gate; int slot = *ws_slot;
  #pragma unroll
  for (int u = 0; u < 4; ++u) {
    size_t i = base + threadIdx.x + u * 256;
    int s = (int)(i >> 8);
    float x = keys[i];
    if (s == slot) x = (1.0f - g) * x + g * ws_wk[i & (KD - 1)];
    out[i] = x;
  }
}

__global__ __launch_bounds__(256) void k_newvalues(const float* __restrict__ values,
                                                   const float* __restrict__ ws_wv,
                                                   const float* __restrict__ ws_gate,
                                                   const int* __restrict__ ws_slot,
                                                   float* __restrict__ out) {
  size_t base = (size_t)blockIdx.x * 1024;
  float g = *ws_gate; int slot = *ws_slot;
  #pragma unroll
  for (int u = 0; u < 4; ++u) {
    size_t i = base + threadIdx.x + u * 256;
    int s = (int)(i >> 10);
    float x = values[i];
    if (s == slot) x = (1.0f - g) * x + g * ws_wv[i & (DH - 1)];
    out[i] = x;
  }
}

// ---------------------------------------------------------------------------
// Host launcher
// ---------------------------------------------------------------------------
extern "C" void kernel_launch(void* const* d_in, const int* in_sizes, int n_in,
                              void* d_out, int out_size, void* d_ws, size_t ws_size,
                              hipStream_t stream) {
  const float*         hidden = (const float*)d_in[0];
  const float*         w_q    = (const float*)d_in[1];
  const float*         b_q    = (const float*)d_in[2];
  const float*         w_g    = (const float*)d_in[3];
  const float*         b_g    = (const float*)d_in[4];
  const float*         keys   = (const float*)d_in[5];
  const float*         values = (const float*)d_in[6];
  const unsigned char* valid  = (const unsigned char*)d_in[7];
  (void)in_sizes; (void)n_in; (void)out_size; (void)ws_size;

  float* out = (float*)d_out;
  // Output layout (flat floats, reference return order):
  float* out_context = out;                                  // 8192*1024
  float* out_margin  = out + (size_t)NROW * DH;              // 1
  float* out_slot    = out_margin + 1;                       // 8192
  float* out_nkeys   = out_slot + NROW;                      // 32768*256
  float* out_nvals   = out_nkeys + (size_t)SK * KD;          // 32768*1024

  // Workspace layout (256B-aligned slices)
  char* base = (char*)d_ws;
  size_t off = 0;
  auto take = [&](size_t bytes) -> char* {
    char* p = base + off;
    off = (off + bytes + 255) & ~(size_t)255;
    return p;
  };
  bf16_t*   q_hi      = (bf16_t*)  take((size_t)NROW * KD * sizeof(bf16_t));
  bf16_t*   q_lo      = (bf16_t*)  take((size_t)NROW * KD * sizeof(bf16_t));
  bf16_t*   kn_hi     = (bf16_t*)  take((size_t)SK * KD * sizeof(bf16_t));
  bf16_t*   kn_lo     = (bf16_t*)  take((size_t)SK * KD * sizeof(bf16_t));
  unsigned* validbits = (unsigned*)take(1024 * sizeof(unsigned));
  int*      anyvalid  = (int*)     take(sizeof(int));
  float*    top_vals  = (float*)   take((size_t)NROW * TOPK * sizeof(float));
  int*      top_idx   = (int*)     take((size_t)NROW * TOPK * sizeof(int));
  float*    g_row     = (float*)   take((size_t)NROW * sizeof(float));
  float*    qcol_part = (float*)   take((size_t)512 * KD * sizeof(float));
  float*    hcol_part = (float*)   take((size_t)64 * DH * sizeof(float));
  float*    ws_gate   = (float*)   take(sizeof(float));
  int*      ws_slot   = (int*)     take(sizeof(int));
  float*    ws_wk     = (float*)   take(KD * sizeof(float));
  float*    ws_wv     = (float*)   take(DH * sizeof(float));

  // 1) valid bitmask + any(valid)
  k_validbits<<<1, 1024, 0, stream>>>(valid, validbits, anyvalid);
  // 2) normalize keys -> bf16 hi/lo
  k_keynorm<<<SK, 256, 0, stream>>>(keys, kn_hi, kn_lo);
  // 3) per-row gate logits
  k_gate<<<NROW, 256, 0, stream>>>(hidden, w_g, b_g, g_row);
  // 4) hidden column partial sums
  k_colsum<<<64, 256, 0, stream>>>(hidden, hcol_part);
  // 5) q projection + normalize (WMMA)
  k_proj<<<NROW / 16, 128, 0, stream>>>(hidden, w_q, b_q, q_hi, q_lo, qcol_part);
  // 6) fused scores GEMM + top-8 (WMMA, no score materialization)
  k_scores_topk<<<NROW / 16, 128, 0, stream>>>(q_hi, q_lo, kn_hi, kn_lo,
                                               validbits, anyvalid,
                                               top_vals, top_idx, out_slot);
  // 7) softmax + value gather -> context
  k_context<<<NROW, 256, 0, stream>>>(top_vals, top_idx, values, out_context);
  // 8) deterministic epilogue scalars/vectors
  k_epilogue<<<1, 256, 0, stream>>>(top_vals, top_idx, g_row, qcol_part,
                                    hcol_part, out_margin, ws_gate, ws_slot,
                                    ws_wk, ws_wv);
  // 9) blended copies
  k_newkeys<<<(SK * KD) / 1024, 256, 0, stream>>>(keys, ws_wk, ws_gate, ws_slot, out_nkeys);
  k_newvalues<<<(SK * DH) / 1024, 256, 0, stream>>>(values, ws_wv, ws_gate, ws_slot, out_nvals);
}
